// DriftAwareMultiHeadAttention_71141838291531
// MI455X (gfx1250) — matrix operand order repaired and checked by reading, not compiled, not run on hardware
//
#include <hip/hip_runtime.h>
#include <hip/hip_bf16.h>

// ---------------------------------------------------------------------------
// DriftAwareMultiHeadAttention on MI455X (gfx1250):
//   x[4,2048,1024] f32 -> qkv GEMM (bf16 WMMA) -> flash attention (bf16 WMMA,
//   f32 softmax) -> out GEMM (bf16 WMMA) + bias -> f32 out.
// All matmuls use v_wmma_f32_16x16x32_bf16 with fragment layouts per
// CDNA5 ISA 7.12.2 (wave32). GEMM wave tile is 32Mx64N (8 wmma / k-step,
// B fragments reused across two A fragments) for 2x arithmetic intensity.
// ---------------------------------------------------------------------------

typedef __attribute__((ext_vector_type(16))) __bf16 bf16x16;
typedef __attribute__((ext_vector_type(8)))  __bf16 bf16x8;
typedef __attribute__((ext_vector_type(8)))  float  f32x8;

#define BATCH 4
#define SEQ   2048
#define DIM   1024
#define HEADS 16
#define HD    64
#define TOK   (BATCH * SEQ)       // 8192

static __device__ __forceinline__ f32x8 wmma_bf16(bf16x16 a, bf16x16 b, f32x8 c) {
  return __builtin_amdgcn_wmma_f32_16x16x32_bf16(false, a, false, b, (short)0, c,
                                                 false, false);
}

// A fragment (16x32 bf16, M x K). ISA 7.12.2: lanes 0-15 row M=lane hold
// K={k0..k0+7, k0+16..k0+23}; lanes 16-31 row M=lane-16 hold K+8 of each half.
static __device__ __forceinline__ bf16x16
load_a_frag(const __bf16* __restrict__ base, int ld, int row0, int k0, int lane) {
  const int hi = lane >> 4;
  const __bf16* p = base + (size_t)(row0 + (lane & 15)) * ld + k0 + hi * 8;
  bf16x8 lo = *(const bf16x8*)(p);
  bf16x8 hp = *(const bf16x8*)(p + 16);
  return __builtin_shufflevector(lo, hp, 0, 1, 2, 3, 4, 5, 6, 7,
                                 8, 9, 10, 11, 12, 13, 14, 15);
}

// B fragment (32x16 bf16, K x N), column data contiguous along K in memory at
// base[col*ld + k]. Lanes 0-15: col N=lane, K={k0..k0+7, k0+16..k0+23};
// lanes 16-31: K+8 of each half.
static __device__ __forceinline__ bf16x16
load_b_frag(const __bf16* __restrict__ base, int ld, int k0, int col0, int lane) {
  const int hi = lane >> 4;
  const __bf16* p = base + (size_t)(col0 + (lane & 15)) * ld + k0 + hi * 8;
  bf16x8 lo = *(const bf16x8*)(p);
  bf16x8 hp = *(const bf16x8*)(p + 16);
  return __builtin_shufflevector(lo, hp, 0, 1, 2, 3, 4, 5, 6, 7,
                                 8, 9, 10, 11, 12, 13, 14, 15);
}

// Row reductions across a 16-lane half-wave (C-tile rows live across lanes
// 0-15 / 16-31; xor masks 1..8 stay within each half).
static __device__ __forceinline__ float rowmax16(float v) {
#pragma unroll
  for (int off = 1; off < 16; off <<= 1) v = fmaxf(v, __shfl_xor(v, off, 32));
  return v;
}
static __device__ __forceinline__ float rowsum16(float v) {
#pragma unroll
  for (int off = 1; off < 16; off <<= 1) v += __shfl_xor(v, off, 32);
  return v;
}

// ---------------------------------------------------------------------------
__global__ void cvt_f32_to_bf16(const float* __restrict__ in,
                                __bf16* __restrict__ out, int n) {
  int i = blockIdx.x * blockDim.x + threadIdx.x;
  int stride = gridDim.x * blockDim.x;
  for (; i < n; i += stride) out[i] = (__bf16)in[i];
}

// ---------------------------------------------------------------------------
// QKV projection: C[8192,3072] = Xbf[8192,1024] * Wqkv^T, scattered into
// Q[bh,n,64], K[bh,n,64], Vt[bh,64,n] as bf16.
// Block: 256 thr = 8 waves; wave tile 32Mx64N; WG tile 256Mx64N.
__global__ __launch_bounds__(256) void gemm_qkv(
    const __bf16* __restrict__ X, const __bf16* __restrict__ W,
    __bf16* __restrict__ Q, __bf16* __restrict__ Kb, __bf16* __restrict__ Vt) {
  const int lane = threadIdx.x & 31, wave = threadIdx.x >> 5;
  const int m0 = blockIdx.x * 256 + wave * 32;
  const int n0 = blockIdx.y * 64;

  f32x8 acc[2][4] = {{f32x8{}, f32x8{}, f32x8{}, f32x8{}},
                     {f32x8{}, f32x8{}, f32x8{}, f32x8{}}};
  for (int k0 = 0; k0 < DIM; k0 += 32) {
    if (k0 + 128 < DIM) {  // gfx1250 global_prefetch_b8, one L2 line ahead
      __builtin_prefetch(X + (size_t)(m0 + (lane & 15)) * DIM + k0 + 128, 0, 0);
      __builtin_prefetch(X + (size_t)(m0 + 16 + (lane & 15)) * DIM + k0 + 128, 0, 0);
    }
    bf16x16 a0 = load_a_frag(X, DIM, m0, k0, lane);
    bf16x16 a1 = load_a_frag(X, DIM, m0 + 16, k0, lane);
#pragma unroll
    for (int t = 0; t < 4; ++t) {
      bf16x16 b = load_b_frag(W, DIM, k0, n0 + t * 16, lane);
      acc[0][t] = wmma_bf16(a0, b, acc[0][t]);
      acc[1][t] = wmma_bf16(a1, b, acc[1][t]);
    }
  }

  const int hi = lane >> 4, c = lane & 15;
#pragma unroll
  for (int mi = 0; mi < 2; ++mi) {
#pragma unroll
    for (int t = 0; t < 4; ++t) {
#pragma unroll
      for (int r = 0; r < 8; ++r) {
        const int m = m0 + mi * 16 + r + hi * 8;
        const int e = n0 + t * 16 + c;
        const __bf16 v = (__bf16)acc[mi][t][r];
        const int b = m >> 11, tok = m & (SEQ - 1);
        if (e < DIM) {                       // Q: [bh, tok, d]
          const int h = e >> 6, d = e & 63;
          Q[(((size_t)(b * HEADS + h) * SEQ) + tok) * HD + d] = v;
        } else if (e < 2 * DIM) {            // K: [bh, tok, d]
          const int e2 = e - DIM, h = e2 >> 6, d = e2 & 63;
          Kb[(((size_t)(b * HEADS + h) * SEQ) + tok) * HD + d] = v;
        } else {                             // V transposed: [bh, d, tok]
          const int e2 = e - 2 * DIM, h = e2 >> 6, d = e2 & 63;
          Vt[((size_t)(b * HEADS + h) * HD + d) * SEQ + tok] = v;
        }
      }
    }
  }
}

// ---------------------------------------------------------------------------
// Fused flash attention. grid = (64 bh, 16), block = 256 (8 waves), each wave
// owns one 16-query tile; keys processed in chunks of 32 with online softmax.
__global__ __launch_bounds__(256) void flash_attn(
    const __bf16* __restrict__ Q, const __bf16* __restrict__ Kb,
    const __bf16* __restrict__ Vt, const float* __restrict__ rescale,
    __bf16* __restrict__ attn) {
  __shared__ __bf16 sP[8][16 * 32];  // per-wave 16x32 P staging (C->A layout)

  const int lane = threadIdx.x & 31, wave = threadIdx.x >> 5;
  const int bh = blockIdx.x, b = bh >> 4, h = bh & 15;
  const int q0 = (blockIdx.y * 8 + wave) * 16;

  const __bf16* Qh = Q  + (size_t)bh * SEQ * HD;
  const __bf16* Kh = Kb + (size_t)bh * SEQ * HD;
  const __bf16* Vh = Vt + (size_t)bh * HD * SEQ;
  const float scale = rescale[0] * 0.125f;  // 1/sqrt(64)

  const bf16x16 qa0 = load_a_frag(Qh, HD, q0, 0, lane);
  const bf16x16 qa1 = load_a_frag(Qh, HD, q0, 32, lane);

  float mrow[8], lrow[8];
  f32x8 O[4] = {f32x8{}, f32x8{}, f32x8{}, f32x8{}};
#pragma unroll
  for (int r = 0; r < 8; ++r) { mrow[r] = -1e30f; lrow[r] = 0.0f; }

  const f32x8 zero{};
  for (int kc = 0; kc < SEQ; kc += 32) {
    if (kc + 64 < SEQ) {  // prefetch next key/value chunk
      __builtin_prefetch(Kh + (size_t)(kc + 64 + (lane & 15)) * HD, 0, 0);
      __builtin_prefetch(Vh + (size_t)(lane & 15) * SEQ + kc + 64, 0, 0);
    }
    // S = Q * K^T for 32 keys (two 16x16 C tiles, K-dim 64 = 2 chained wmma)
    f32x8 s0 = wmma_bf16(qa1, load_b_frag(Kh, HD, 32, kc, lane),
               wmma_bf16(qa0, load_b_frag(Kh, HD, 0,  kc, lane), zero));
    f32x8 s1 = wmma_bf16(qa1, load_b_frag(Kh, HD, 32, kc + 16, lane),
               wmma_bf16(qa0, load_b_frag(Kh, HD, 0,  kc + 16, lane), zero));

    __syncthreads();  // previous iteration's P reads complete before overwrite
#pragma unroll
    for (int r = 0; r < 8; ++r) {
      const float a0 = s0[r] * scale, a1 = s1[r] * scale;
      const float tmax = rowmax16(fmaxf(a0, a1));
      const float mn = fmaxf(mrow[r], tmax);
      const float corr = __expf(mrow[r] - mn);
      const float p0 = __expf(a0 - mn), p1 = __expf(a1 - mn);
      lrow[r] = lrow[r] * corr + rowsum16(p0 + p1);
      mrow[r] = mn;
      O[0][r] *= corr; O[1][r] *= corr; O[2][r] *= corr; O[3][r] *= corr;
      const int rowm = r + ((lane >> 4) << 3);
      sP[wave][rowm * 32 + (lane & 15)]      = (__bf16)p0;
      sP[wave][rowm * 32 + 16 + (lane & 15)] = (__bf16)p1;
    }
    __syncthreads();  // P visible in LDS

    // Reload P in A-fragment layout, then O += P * V (4 dim tiles of 16)
    const int hi = lane >> 4;
    const __bf16* pp = &sP[wave][(lane & 15) * 32 + hi * 8];
    bf16x8 plo = *(const bf16x8*)(pp);
    bf16x8 phi = *(const bf16x8*)(pp + 16);
    bf16x16 pa = __builtin_shufflevector(plo, phi, 0, 1, 2, 3, 4, 5, 6, 7,
                                         8, 9, 10, 11, 12, 13, 14, 15);
#pragma unroll
    for (int t = 0; t < 4; ++t)
      O[t] = wmma_bf16(pa, load_b_frag(Vh, SEQ, kc, t * 16, lane), O[t]);
  }

  const int hi = lane >> 4, c = lane & 15;
#pragma unroll
  for (int r = 0; r < 8; ++r) {
    const float inv = 1.0f / lrow[r];
    const int qrow = q0 + r + hi * 8;
    const size_t base = ((size_t)(b * SEQ + qrow)) * DIM + h * HD;
#pragma unroll
    for (int t = 0; t < 4; ++t)
      attn[base + t * 16 + c] = (__bf16)(O[t][r] * inv);
  }
}

// ---------------------------------------------------------------------------
// Output projection: out[8192,1024] = attn_bf[8192,1024] * Wout^T + bias (f32)
__global__ __launch_bounds__(256) void gemm_out(
    const __bf16* __restrict__ A, const __bf16* __restrict__ W,
    const float* __restrict__ bias, float* __restrict__ out) {
  const int lane = threadIdx.x & 31, wave = threadIdx.x >> 5;
  const int m0 = blockIdx.x * 256 + wave * 32;
  const int n0 = blockIdx.y * 64;

  f32x8 acc[2][4] = {{f32x8{}, f32x8{}, f32x8{}, f32x8{}},
                     {f32x8{}, f32x8{}, f32x8{}, f32x8{}}};
  for (int k0 = 0; k0 < DIM; k0 += 32) {
    if (k0 + 128 < DIM) {
      __builtin_prefetch(A + (size_t)(m0 + (lane & 15)) * DIM + k0 + 128, 0, 0);
      __builtin_prefetch(A + (size_t)(m0 + 16 + (lane & 15)) * DIM + k0 + 128, 0, 0);
    }
    bf16x16 a0 = load_a_frag(A, DIM, m0, k0, lane);
    bf16x16 a1 = load_a_frag(A, DIM, m0 + 16, k0, lane);
#pragma unroll
    for (int t = 0; t < 4; ++t) {
      bf16x16 b = load_b_frag(W, DIM, k0, n0 + t * 16, lane);
      acc[0][t] = wmma_bf16(a0, b, acc[0][t]);
      acc[1][t] = wmma_bf16(a1, b, acc[1][t]);
    }
  }

  const int hi = lane >> 4, c = lane & 15;
#pragma unroll
  for (int mi = 0; mi < 2; ++mi) {
#pragma unroll
    for (int t = 0; t < 4; ++t) {
#pragma unroll
      for (int r = 0; r < 8; ++r) {
        const int m = m0 + mi * 16 + r + hi * 8;
        const int e = n0 + t * 16 + c;
        out[(size_t)m * DIM + e] = acc[mi][t][r] + bias[e];
      }
    }
  }
}

// ---------------------------------------------------------------------------
extern "C" void kernel_launch(void* const* d_in, const int* in_sizes, int n_in,
                              void* d_out, int out_size, void* d_ws, size_t ws_size,
                              hipStream_t stream) {
  (void)in_sizes; (void)n_in; (void)out_size; (void)ws_size;

  const float* x       = (const float*)d_in[0];  // [4,2048,1024]
  const float* w_qkv   = (const float*)d_in[1];  // [3072,1024]
  const float* w_out   = (const float*)d_in[2];  // [1024,1024]
  const float* b_out   = (const float*)d_in[3];  // [1024]
  const float* rescale = (const float*)d_in[4];  // [1]
  float* out = (float*)d_out;

  char* ws = (char*)d_ws;
  __bf16* x_bf    = (__bf16*)(ws);                       // 16 MB
  __bf16* wqkv_bf = (__bf16*)(ws + (size_t)16777216);    //  6 MB
  __bf16* wout_bf = (__bf16*)(ws + (size_t)23068672);    //  2 MB
  __bf16* Qbuf    = (__bf16*)(ws + (size_t)25165824);    // 16 MB  [bh,n,64]
  __bf16* Kbuf    = (__bf16*)(ws + (size_t)41943040);    // 16 MB  [bh,n,64]
  __bf16* Vtbuf   = (__bf16*)(ws + (size_t)58720256);    // 16 MB  [bh,64,n]
  __bf16* attn    = (__bf16*)(ws + (size_t)75497472);    // 16 MB  [tok,1024]

  cvt_f32_to_bf16<<<2048, 256, 0, stream>>>(x, x_bf, TOK * DIM);
  cvt_f32_to_bf16<<<2048, 256, 0, stream>>>(w_qkv, wqkv_bf, 3 * DIM * DIM);
  cvt_f32_to_bf16<<<1024, 256, 0, stream>>>(w_out, wout_bf, DIM * DIM);

  gemm_qkv<<<dim3(TOK / 256, (3 * DIM) / 64), 256, 0, stream>>>(
      x_bf, wqkv_bf, Qbuf, Kbuf, Vtbuf);

  flash_attn<<<dim3(BATCH * HEADS, SEQ / (16 * 8)), 256, 0, stream>>>(
      Qbuf, Kbuf, Vtbuf, rescale, attn);

  gemm_out<<<dim3(TOK / 256, DIM / 64), 256, 0, stream>>>(
      attn, wout_bf, b_out, out);
}
